// CLIP_77232101917117
// MI455X (gfx1250) — matrix operand order repaired and checked by reading, not compiled, not run on hardware
//
#include <hip/hip_runtime.h>
#include <hip/hip_bf16.h>

// ---------------- problem constants ----------------
constexpr int N_NODES = 50000;
constexpr int DIM     = 256;    // input channels
constexpr int HEADS   = 4;
constexpr int CH      = 128;    // out channels per head
constexpr int HC      = HEADS * CH;        // 512
constexpr int NPAD    = ((N_NODES + 63) / 64) * 64;  // 50048

typedef __attribute__((ext_vector_type(16))) __bf16 v16bf;
typedef __attribute__((ext_vector_type(8)))  float  v8f;
typedef __attribute__((ext_vector_type(4)))  unsigned u32x4;
typedef __attribute__((ext_vector_type(8)))  unsigned u32x8;
typedef __attribute__((ext_vector_type(4)))  int      i32x4;

#if defined(__gfx1250__)
using as3_void = __attribute__((address_space(3))) void;
typedef __attribute__((address_space(1))) i32x4 as1_i32x4;
typedef __attribute__((address_space(3))) i32x4 as3_i32x4;
#if __has_builtin(__builtin_amdgcn_global_load_async_to_lds_b128) && \
    __has_builtin(__builtin_amdgcn_s_wait_asynccnt)
#define USE_ASYNC_LDS 1
#else
#define USE_ASYNC_LDS 0
#endif
#else
#define USE_ASYNC_LDS 0
#endif

// per-lane 16B global->LDS stage: async-to-LDS (ASYNCcnt) when available
__device__ __forceinline__ void stage_b128(void* lds_dst, const void* gsrc) {
#if USE_ASYNC_LDS
  __builtin_amdgcn_global_load_async_to_lds_b128(
      (as1_i32x4*)gsrc, (as3_i32x4*)lds_dst, /*offset=*/0, /*cpol=*/0);
#else
  *(uint4*)lds_dst = *(const uint4*)gsrc;
#endif
}
__device__ __forceinline__ void stage_wait() {
#if USE_ASYNC_LDS
  __builtin_amdgcn_s_wait_asynccnt(0);
#endif
}

// ---------------- WMMA helpers ----------------
__device__ __forceinline__ v8f wmma_bf16(v16bf a, v16bf b, v8f c) {
  // D = A(16x32) x B(32x16) + C, f32 accumulate
  return __builtin_amdgcn_wmma_f32_16x16x32_bf16(
      /*neg_a=*/false, a, /*neg_b=*/false, b,
      /*c_mod=*/(short)0, c, /*reuse_a=*/false, /*reuse_b=*/false);
}

// A fragment: tile row-major [row][32]; lane L<16 -> row L, K in {0..15};
// lane>=16 -> row L-16, K in {16..31}.  Two 16B LDS loads.
__device__ __forceinline__ v16bf frag_A(const __bf16* rowbase, int lane) {
  const uint4* p = (const uint4*)rowbase;      // 32 bf16 = 4 x uint4
  int i0 = (lane & 16) ? 2 : 0;
  union { uint4 q[2]; v16bf v; } u;
  u.q[0] = p[i0];
  u.q[1] = p[i0 + 1];
  return u.v;
}

// B fragment from transposed tile [col][32]; lane col = lane&15,
// K = {0..15} for lanes 0-15, {16..31} for lanes 16-31.  Two 16B LDS loads.
__device__ __forceinline__ v16bf frag_B(const __bf16* colbase, int lane) {
  const uint4* p = (const uint4*)colbase;
  int i0 = (lane & 16) ? 2 : 0;
  union { uint4 q[2]; v16bf v; } u;
  u.q[0] = p[i0];
  u.q[1] = p[i0 + 1];
  return u.v;
}

// ---------------- K0: zero small accumulators ----------------
__global__ void sgf_zero(float* p, int n) {
  int i = blockIdx.x * blockDim.x + threadIdx.x;
  if (i < n) p[i] = 0.0f;
}

// ---------------- K1: f32 -> bf16 conversion of x and Wq/Wk/Wv ----------------
__global__ void sgf_convert(const float* __restrict__ x,
                            const float* __restrict__ Wq,
                            const float* __restrict__ Wk,
                            const float* __restrict__ Wv,
                            __bf16* __restrict__ xb,
                            __bf16* __restrict__ wqb,
                            __bf16* __restrict__ wkb,
                            __bf16* __restrict__ wvb) {
  const long total_x = (long)NPAD * DIM;
  const long wsz     = (long)DIM * HC;
  long i = (long)blockIdx.x * blockDim.x + threadIdx.x;
  if (i < total_x) {
    xb[i] = (i < (long)N_NODES * DIM) ? (__bf16)x[i] : (__bf16)0.0f;
  } else {
    long j = i - total_x;
    if (j < wsz)            wqb[j]           = (__bf16)Wq[j];
    else if (j < 2 * wsz)   wkb[j - wsz]     = (__bf16)Wk[j - wsz];
    else if (j < 3 * wsz)   wvb[j - 2 * wsz] = (__bf16)Wv[j - 2 * wsz];
  }
}

// ---------------- K2: fused QKV GEMM: out = x@W + b, bf16 store, + sum-of-squares ----
// grid: (NPAD/64, HC/64, 3)   block: 256 (8 waves), wave = 16x32 band (2 wmma accs)
// A (x) tile staged by the Tensor Data Mover (one 2D-tile DMA per K-step).
__global__ void sgf_qkv_gemm(const __bf16* __restrict__ xb,
                             const __bf16* __restrict__ wqb,
                             const __bf16* __restrict__ wkb,
                             const __bf16* __restrict__ wvb,
                             const float* __restrict__ bq,
                             const float* __restrict__ bk,
                             const float* __restrict__ bv,
                             __bf16* __restrict__ qb,
                             __bf16* __restrict__ kb,
                             __bf16* __restrict__ vb,
                             float* __restrict__ norms) {
  __shared__ __align__(16) __bf16 sA[64 * 32];   // [row][k]
  __shared__ __align__(16) __bf16 sB[64 * 32];   // [col][k] (transposed)

  const int z    = blockIdx.z;
  const __bf16* wb   = (z == 0) ? wqb : (z == 1) ? wkb : wvb;
  const float*  bias = (z == 0) ? bq  : (z == 1) ? bk  : bv;
  __bf16*       outp = (z == 0) ? qb  : (z == 1) ? kb  : vb;

  const int tid  = threadIdx.x;
  const int lane = tid & 31;
  const int wave = tid >> 5;
  const int waveM = wave >> 1;       // 0..3 -> 16-row band
  const int waveN = wave & 1;        // 0..1 -> 32-col band
  const int row0 = blockIdx.x * 64;
  const int col0 = blockIdx.y * 64;

  v8f acc0 = {}; v8f acc1 = {};

#if defined(__gfx1250__)
  // Tensor DMA descriptor group 1 (loop-invariant): data_size=2B,
  // tensor = DIM x N_NODES (OOB rows auto-zero), tile = 32 x 64, stride0 = DIM.
  u32x8 g1;
  g1[0] = 1u << 16;                                             // data_size=1 (2B)
  g1[1] = ((unsigned)DIM & 0xffffu) << 16;                      // tensor_dim0 lo
  g1[2] = (((unsigned)DIM >> 16) & 0xffffu) |
          (((unsigned)N_NODES & 0xffffu) << 16);                // dim0 hi | dim1 lo
  g1[3] = (((unsigned)N_NODES >> 16) & 0xffffu) | (32u << 16);  // dim1 hi | tile_dim0
  g1[4] = 64u;                                                  // tile_dim1
  g1[5] = (unsigned)DIM;                                        // tensor_dim0_stride lo
  g1[6] = 0u;
  g1[7] = 0u;
  const unsigned ldsA = (unsigned)(size_t)(as3_void*)sA;
#endif

  for (int kk = 0; kk < DIM; kk += 32) {
#if defined(__gfx1250__)
    if (wave == 0) {  // one TDM 2D-tile load per K-step: x[row0:+64, kk:+32] -> sA
      unsigned long long ga =
          (unsigned long long)(size_t)(xb + (size_t)row0 * DIM + kk);
      u32x4 g0;
      g0[0] = 1u;                                               // count=1, user mode
      g0[1] = ldsA;                                             // lds_addr
      g0[2] = (unsigned)ga;                                     // global_addr lo
      g0[3] = (unsigned)((ga >> 32) & 0x01ffffffull) | (2u << 30);  // addr hi | type=2
      asm volatile("tensor_load_to_lds %0, %1" :: "s"(g0), "s"(g1) : "memory");
    }
#else
    {  // non-gfx1250 fallback (host parse only)
      int r = tid >> 2, c = (tid & 3) * 8;
      uint4 val = {};
      if (row0 + r < N_NODES)
        val = *(const uint4*)(xb + (long)(row0 + r) * DIM + kk + c);
      *(uint4*)(sA + r * 32 + c) = val;
    }
#endif
    // stage B transposed: W[k][col] -> sB[col][k]
    {
      int kIdx = tid >> 3, cIdx = (tid & 7) * 8;
      union { uint4 q; __bf16 h[8]; } u;
      u.q = *(const uint4*)(wb + (long)(kk + kIdx) * HC + col0 + cIdx);
#pragma unroll
      for (int j = 0; j < 8; ++j) sB[(cIdx + j) * 32 + kIdx] = u.h[j];
    }
#if defined(__gfx1250__)
    if (wave == 0) __builtin_amdgcn_s_wait_tensorcnt(0);
#endif
    __syncthreads();

    v16bf a  = frag_A(sA + (waveM * 16 + (lane & 15)) * 32, lane);
    v16bf b0 = frag_B(sB + (waveN * 32 + 0  + (lane & 15)) * 32, lane);
    v16bf b1 = frag_B(sB + (waveN * 32 + 16 + (lane & 15)) * 32, lane);
    acc0 = wmma_bf16(a, b0, acc0);
    acc1 = wmma_bf16(a, b1, acc1);
    __syncthreads();
  }

  // epilogue: +bias, bf16 store, sum-of-squares for q/k
  float ssq = 0.0f;
  const int rbase = row0 + waveM * 16 + ((lane & 16) ? 8 : 0);
#pragma unroll
  for (int sub = 0; sub < 2; ++sub) {
    v8f acc = sub ? acc1 : acc0;
    int col = col0 + waveN * 32 + sub * 16 + (lane & 15);
    float bcol = bias[col];
#pragma unroll
    for (int r = 0; r < 8; ++r) {
      int row = rbase + r;
      float v = acc[r] + bcol;
      if (row < N_NODES) {
        outp[(long)row * HC + col] = (__bf16)v;
        ssq += v * v;
      }
    }
  }
  if (z < 2) {
#pragma unroll
    for (int m = 16; m >= 1; m >>= 1) ssq += __shfl_xor(ssq, m, 32);
    if (lane == 0) atomicAdd(&norms[z], ssq);
  }
}

// ---------------- K3s: ks_sum[col] = sum_l k[l][col] ----------------
__global__ void sgf_ksum(const __bf16* __restrict__ kb, float* __restrict__ ks_sum) {
  int col = blockIdx.x * 256 + threadIdx.x;
  const int chunk = (N_NODES + 63) / 64;  // 782
  int l0 = blockIdx.y * chunk;
  int l1 = min(N_NODES, l0 + chunk);
  float s = 0.0f;
  for (int l = l0; l < l1; ++l) s += (float)kb[(long)l * HC + col];
  atomicAdd(&ks_sum[col], s);
}

// ---------------- K3: kvs[h] += K_h^T V_h over an L chunk (split-K atomics) ----
// grid: (L_chunks, 4 tiles, HEADS)  block 256.  Both stagings are transposes,
// so they stay per-element; prefetch the next L-slab while computing.
__global__ void sgf_kvs(const __bf16* __restrict__ kb,
                        const __bf16* __restrict__ vb,
                        float* __restrict__ kvs) {
  __shared__ __align__(16) __bf16 sKA[64 * 32];  // [m][l]  (A rows = k-columns)
  __shared__ __align__(16) __bf16 sVB[64 * 32];  // [col][l]

  constexpr int CHUNK = 2048;
  const int tid  = threadIdx.x;
  const int lane = tid & 31;
  const int wave = tid >> 5;
  const int waveM = wave >> 1, waveN = wave & 1;
  const int h   = blockIdx.z, hbase = h * CH;
  const int tm  = (blockIdx.y >> 1) * 64;
  const int tn  = (blockIdx.y & 1) * 64;
  const int lstart = blockIdx.x * CHUNK;

  v8f acc0 = {}; v8f acc1 = {};

  for (int l0 = lstart; l0 < lstart + CHUNK; l0 += 32) {
    int lIdx = tid >> 3, cIdx = (tid & 7) * 8;
    bool inb = (l0 + lIdx) < N_NODES;
    // stage Kᵀ: k[l][hbase+tm+m] -> sKA[m][l]
    {
      union { uint4 q; __bf16 hv[8]; } u; u.q = uint4{};
      if (inb) u.q = *(const uint4*)(kb + (long)(l0 + lIdx) * HC + hbase + tm + cIdx);
#pragma unroll
      for (int j = 0; j < 8; ++j) sKA[(cIdx + j) * 32 + lIdx] = u.hv[j];
    }
    // stage V: v[l][hbase+tn+c] -> sVB[c][l]
    {
      union { uint4 q; __bf16 hv[8]; } u; u.q = uint4{};
      if (inb) u.q = *(const uint4*)(vb + (long)(l0 + lIdx) * HC + hbase + tn + cIdx);
#pragma unroll
      for (int j = 0; j < 8; ++j) sVB[(cIdx + j) * 32 + lIdx] = u.hv[j];
    }
    if (inb && l0 + 32 + lIdx < N_NODES) {
      __builtin_prefetch(kb + (long)(l0 + 32 + lIdx) * HC + hbase + tm + cIdx, 0, 0);
      __builtin_prefetch(vb + (long)(l0 + 32 + lIdx) * HC + hbase + tn + cIdx, 0, 0);
    }
    __syncthreads();

    v16bf a  = frag_A(sKA + (waveM * 16 + (lane & 15)) * 32, lane);
    v16bf b0 = frag_B(sVB + (waveN * 32 + 0  + (lane & 15)) * 32, lane);
    v16bf b1 = frag_B(sVB + (waveN * 32 + 16 + (lane & 15)) * 32, lane);
    acc0 = wmma_bf16(a, b0, acc0);
    acc1 = wmma_bf16(a, b1, acc1);
    __syncthreads();
  }

  const int rbase = tm + waveM * 16 + ((lane & 16) ? 8 : 0);
#pragma unroll
  for (int sub = 0; sub < 2; ++sub) {
    v8f acc = sub ? acc1 : acc0;
    int col = tn + waveN * 32 + sub * 16 + (lane & 15);
#pragma unroll
    for (int r = 0; r < 8; ++r)
      atomicAdd(&kvs[(long)h * CH * CH + (long)(rbase + r) * CH + col], acc[r]);
  }
}

// ---------------- K3b: nd[n,h] = dot(q[n,h,:], ks_sum[h,:]) ----------------
__global__ void sgf_nd(const __bf16* __restrict__ qb,
                       const float* __restrict__ ks_sum,
                       float* __restrict__ nd) {
  int t = blockIdx.x * blockDim.x + threadIdx.x;
  if (t >= N_NODES * HEADS) return;
  int n = t >> 2, h = t & 3;
  const __bf16* qrow = qb + (long)n * HC + h * CH;
  const float*  ks   = ks_sum + h * CH;
  float s = 0.0f;
#pragma unroll 8
  for (int c = 0; c < CH; ++c) s += (float)qrow[c] * ks[c];
  nd[t] = s;
}

// ---------------- K4: final fused q@kvs + n*v, normalize, mean heads -------
// grid: (NPAD/64, CH/64)  block 256; loops heads internally, writes d_out f32.
// q tile staged via async global->LDS (ASYNCcnt) when the builtin exists.
__global__ void sgf_final(const __bf16* __restrict__ qb,
                          const __bf16* __restrict__ vb,
                          const float* __restrict__ kvs,
                          const float* __restrict__ nd,
                          const float* __restrict__ norms,
                          float* __restrict__ out) {
  __shared__ __align__(16) __bf16 sA[64 * 32];   // q tile [row][k]
  __shared__ __align__(16) __bf16 sB[64 * 32];   // kvs tile [col][k]

  const int tid  = threadIdx.x;
  const int lane = tid & 31;
  const int wave = tid >> 5;
  const int waveM = wave >> 1, waveN = wave & 1;
  const int row0  = blockIdx.x * 64;
  const int colC0 = blockIdx.y * 64;
  const float Nf = (float)N_NODES;
  const float s  = 1.0f / (sqrtf(norms[0]) * sqrtf(norms[1]));  // fold 1/(||q|| ||k||)

  v8f outAcc0 = {}; v8f outAcc1 = {};
  const int rbase = row0 + waveM * 16 + ((lane & 16) ? 8 : 0);

  for (int h = 0; h < HEADS; ++h) {
    v8f acc0 = {}; v8f acc1 = {};
    for (int kk = 0; kk < CH; kk += 32) {
      {  // stage q: async global->LDS per lane (16B each)
        int r = tid >> 2, c = (tid & 3) * 8;
        if (row0 + r < N_NODES)
          stage_b128(sA + r * 32 + c,
                     qb + (long)(row0 + r) * HC + h * CH + kk + c);
        else
          *(uint4*)(sA + r * 32 + c) = uint4{};
      }
      {  // stage kvs (f32 -> bf16, transposed)
        int kIdx = tid >> 3, cIdx = (tid & 7) * 8;
        const float* src = kvs + (long)h * CH * CH + (long)(kk + kIdx) * CH + colC0 + cIdx;
#pragma unroll
        for (int j = 0; j < 8; ++j) sB[(cIdx + j) * 32 + kIdx] = (__bf16)src[j];
      }
      stage_wait();
      __syncthreads();
      v16bf a  = frag_A(sA + (waveM * 16 + (lane & 15)) * 32, lane);
      v16bf b0 = frag_B(sB + (waveN * 32 + 0  + (lane & 15)) * 32, lane);
      v16bf b1 = frag_B(sB + (waveN * 32 + 16 + (lane & 15)) * 32, lane);
      acc0 = wmma_bf16(a, b0, acc0);
      acc1 = wmma_bf16(a, b1, acc1);
      __syncthreads();
    }
    // combine this head: (s*acc + N*v) / (s*nd + N), averaged over heads
    float invden[8];
#pragma unroll
    for (int r = 0; r < 8; ++r) {
      float den = s * nd[(long)(rbase + r) * HEADS + h] + Nf;
      invden[r] = 0.25f / den;
    }
#pragma unroll
    for (int sub = 0; sub < 2; ++sub) {
      v8f acc = sub ? acc1 : acc0;
      int col = colC0 + waveN * 32 + sub * 16 + (lane & 15);
#pragma unroll
      for (int r = 0; r < 8; ++r) {
        int row = rbase + r;
        float vf = (row < N_NODES) ? (float)vb[(long)row * HC + h * CH + col] : 0.0f;
        float contrib = (s * acc[r] + Nf * vf) * invden[r];
        if (sub == 0) outAcc0[r] += contrib; else outAcc1[r] += contrib;
      }
    }
  }

#pragma unroll
  for (int sub = 0; sub < 2; ++sub) {
    v8f oa = sub ? outAcc1 : outAcc0;
    int col = colC0 + waveN * 32 + sub * 16 + (lane & 15);
#pragma unroll
    for (int r = 0; r < 8; ++r) {
      int row = rbase + r;
      if (row < N_NODES) out[(long)row * CH + col] = oa[r];
    }
  }
}

// ---------------- host side ----------------
extern "C" void kernel_launch(void* const* d_in, const int* in_sizes, int n_in,
                              void* d_out, int out_size, void* d_ws, size_t ws_size,
                              hipStream_t stream) {
  (void)in_sizes; (void)n_in; (void)out_size; (void)ws_size;
  const float* x  = (const float*)d_in[0];
  const float* Wq = (const float*)d_in[1];
  const float* bq = (const float*)d_in[2];
  const float* Wk = (const float*)d_in[3];
  const float* bk = (const float*)d_in[4];
  const float* Wv = (const float*)d_in[5];
  const float* bv = (const float*)d_in[6];
  float* out = (float*)d_out;

  char* base = (char*)d_ws;
  size_t off = 0;
  auto alloc = [&](size_t bytes) { size_t o = off; off = (off + bytes + 255) & ~(size_t)255; return o; };

  const size_t wbytes = (size_t)DIM * HC * 2;
  __bf16* xb  = (__bf16*)(base + alloc((size_t)NPAD * DIM * 2));
  __bf16* wqb = (__bf16*)(base + alloc(wbytes));
  __bf16* wkb = (__bf16*)(base + alloc(wbytes));
  __bf16* wvb = (__bf16*)(base + alloc(wbytes));
  __bf16* qb  = (__bf16*)(base + alloc((size_t)NPAD * HC * 2));
  __bf16* kb  = (__bf16*)(base + alloc((size_t)NPAD * HC * 2));
  __bf16* vb  = (__bf16*)(base + alloc((size_t)NPAD * HC * 2));
  // keep kvs / ks_sum / norms contiguous so one zero-kernel covers them
  float* kvs    = (float*)(base + alloc((size_t)HEADS * CH * CH * 4));
  float* ks_sum = (float*)(base + alloc((size_t)HC * 4));
  float* norms  = (float*)(base + alloc(256));
  float* nd     = (float*)(base + alloc((size_t)NPAD * HEADS * 4));

  const int zero_floats = HEADS * CH * CH + HC + 64;  // kvs + ks_sum + norms pad
  sgf_zero<<<(zero_floats + 255) / 256, 256, 0, stream>>>(kvs, zero_floats);

  const long conv_total = (long)NPAD * DIM + 3L * DIM * HC;
  sgf_convert<<<(unsigned)((conv_total + 255) / 256), 256, 0, stream>>>(
      x, Wq, Wk, Wv, xb, wqb, wkb, wvb);

  dim3 gQKV(NPAD / 64, HC / 64, 3);
  sgf_qkv_gemm<<<gQKV, 256, 0, stream>>>(xb, wqb, wkb, wvb, bq, bk, bv,
                                         qb, kb, vb, norms);

  sgf_ksum<<<dim3(2, 64), 256, 0, stream>>>(kb, ks_sum);

  dim3 gKVS((N_NODES + 2047) / 2048, 4, HEADS);
  sgf_kvs<<<gKVS, 256, 0, stream>>>(kb, vb, kvs);

  sgf_nd<<<(N_NODES * HEADS + 255) / 256, 256, 0, stream>>>(qb, ks_sum, nd);

  dim3 gFin(NPAD / 64, CH / 64);
  sgf_final<<<gFin, 256, 0, stream>>>(qb, vb, kvs, nd, norms, out);
}
